// HeteroGNN_2783138807968
// MI455X (gfx1250) — matrix operand order repaired and checked by reading, not compiled, not run on hardware
//
#include <hip/hip_runtime.h>
#include <hip/hip_bf16.h>
#include <math.h>

// ---------------------------------------------------------------------------
// MI455X (gfx1250) HeteroGNN forward.
// GEMMs: v_wmma_f32_16x16x32_f16, f32 accumulate, 64-row M-macrotile per block
// (4 WMMAs per wave per K-step), W tiles staged into LDS by the Tensor Data
// Mover (tensor_load_to_lds + s_wait_tensorcnt). Edge softmax/scatter are
// HBM/atomic bound (~4 GB traffic -> ~0.2 ms @ 23.3 TB/s).
// ---------------------------------------------------------------------------

typedef __attribute__((ext_vector_type(16))) _Float16 v16h;
typedef __attribute__((ext_vector_type(8)))  _Float16 v8h;
typedef __attribute__((ext_vector_type(8)))  float    v8f;
typedef __attribute__((ext_vector_type(4)))  unsigned v4u;
typedef __attribute__((ext_vector_type(8)))  int      v8i;
typedef __attribute__((ext_vector_type(4)))  int      v4i;

#define HID 128
#define NHEAD 4
#define CHAN 32
#define NEG_SLOPE 0.2f
#define BN_EPS 1e-5f

// ---------------------------------------------------------------------------
// Pre-pass: W [K x NC] fp32 row-major  ->  Wh [NC x K] f16 (transposed),
// so each (col, k-tile) is 64 contiguous bytes for the TDM 2-D descriptor.
// ---------------------------------------------------------------------------
__global__ void w_to_f16t_kernel(const float* __restrict__ W,
                                 _Float16* __restrict__ Wh, int K, int NC) {
    int g = blockIdx.x * blockDim.x + threadIdx.x;
    if (g >= K * NC) return;
    int k = g / NC, n = g % NC;                 // coalesced read of W
    Wh[(size_t)n * K + k] = (_Float16)W[g];
}

// ---------------------------------------------------------------------------
// WMMA GEMM: Y[N, NC] = act(X[N, K] @ W[K, NC] + bias)
// Grid: ceil(N/64) blocks.  Block: NC/16 waves of 32, each wave owns a
// 64x16 strip (4 M-tiles), K loop in steps of 32.
// W tile [NC][32] f16 is DMA'd into LDS by the TDM from the pre-transposed Wh.
// ---------------------------------------------------------------------------
template <int NC>
__global__ void lin_wmma_kernel(const float* __restrict__ X,
                                const _Float16* __restrict__ Wh,  // [NC][K]
                                const float* __restrict__ bias,
                                float* __restrict__ Y,
                                int N, int K, int relu) {
    __shared__ _Float16 sA[64][32];    // X macrotile (f16)
    __shared__ _Float16 sWt[NC][32];   // W tile, col-major by n (TDM layout)

    const int t    = threadIdx.x;
    const int lane = t & 31;
    const int wave = t >> 5;           // N-tile index, 0..NC/16-1
    const int row0 = blockIdx.x * 64;
    const int rr   = lane & 15;
    const int hh   = lane >> 4;

    v8f acc0 = {}, acc1 = {}, acc2 = {}, acc3 = {};

    for (int k0 = 0; k0 < K; k0 += 32) {
        __syncthreads();   // previous tile fully consumed
        // cooperative load X macrotile (64x32 fp32 -> f16), row-clamped
        for (int i = t; i < 64 * 32; i += blockDim.x) {
            int r = i >> 5, c = i & 31;
            int gr = row0 + r;
            if (gr >= N) gr = N - 1;
            sA[r][c] = (_Float16)X[(size_t)gr * K + (k0 + c)];
        }
        if (k0 + 32 < K) {
            __builtin_prefetch(&X[(size_t)((row0 + (t & 63)) < N ? (row0 + (t & 63)) : (N - 1)) * K + k0 + 32], 0, 0);
        }
        if (wave == 0) {
            // --- Tensor Data Mover: DMA W tile [NC rows x 32 halves] -> LDS ---
            unsigned lds_base = (unsigned)(uintptr_t)&sWt[0][0];   // LDS byte offset
            unsigned long long ga =
                (unsigned long long)(uintptr_t)Wh + (unsigned long long)k0 * 2ull;
            v4u g0;
            g0.x = 1u;                                             // count=1 (valid D#)
            g0.y = lds_base;                                       // lds_addr
            g0.z = (unsigned)(ga & 0xffffffffu);                   // global_addr[31:0]
            g0.w = (unsigned)((ga >> 32) & 0x01ffffffu) | (2u << 30); // ga[56:32], type=2
            v8i g1;
            g1[0] = (int)(1u << 16);                               // data_size=2B
            g1[1] = (int)(((unsigned)K & 0xffffu) << 16);          // tensor_dim0[15:0]
            g1[2] = (int)((((unsigned)K >> 16) & 0xffffu) |
                          (((unsigned)NC & 0xffffu) << 16));       // dim0 hi | dim1 lo
            g1[3] = (int)(32u << 16);                              // dim1 hi=0 | tile_dim0=32
            g1[4] = (int)((unsigned)NC & 0xffffu);                 // tile_dim1=NC, tile_dim2=0
            g1[5] = (int)(unsigned)K;                              // dim0_stride[31:0]=K
            g1[6] = 0;                                             // stride hi / dim1_stride lo
            g1[7] = 0;
            v4i gz = {0, 0, 0, 0};
            v8i gz8 = {0, 0, 0, 0, 0, 0, 0, 0};
            __builtin_amdgcn_tensor_load_to_lds(g0, g1, gz, gz, gz8, 0);
            __builtin_amdgcn_s_wait_tensorcnt(0);
        }
        __syncthreads();   // X stores visible + W tile complete for all waves

        // B fragment: lane col (wave*16+rr); halves K = 16h..16h+15
        int col = wave * 16 + rr;
        v8h b0 = *(const v8h*)&sWt[col][16 * hh];
        v8h b1 = *(const v8h*)&sWt[col][16 * hh + 8];
        v16h b = __builtin_shufflevector(b0, b1, 0, 1, 2, 3, 4, 5, 6, 7,
                                         8, 9, 10, 11, 12, 13, 14, 15);
#pragma unroll
        for (int m = 0; m < 4; ++m) {
            // A fragment: lane row m*16+rr; halves K = 16h..16h+15
            v8h a0 = *(const v8h*)&sA[m * 16 + rr][16 * hh];
            v8h a1 = *(const v8h*)&sA[m * 16 + rr][16 * hh + 8];
            v16h a = __builtin_shufflevector(a0, a1, 0, 1, 2, 3, 4, 5, 6, 7,
                                             8, 9, 10, 11, 12, 13, 14, 15);
            v8f* pa = (m == 0) ? &acc0 : (m == 1) ? &acc1 : (m == 2) ? &acc2 : &acc3;
            *pa = __builtin_amdgcn_wmma_f32_16x16x32_f16(
                false, a, false, b, (short)0, *pa, false, false);
        }
    }

    // D layout: lane col = wave*16 + rr, rows m*16 + 8*hh + j
    int col = wave * 16 + rr;
    float bv = bias ? bias[col] : 0.0f;
#pragma unroll
    for (int m = 0; m < 4; ++m) {
        v8f acc = (m == 0) ? acc0 : (m == 1) ? acc1 : (m == 2) ? acc2 : acc3;
#pragma unroll
        for (int j = 0; j < 8; ++j) {
            int grow = row0 + m * 16 + 8 * hh + j;
            if (grow < N) {
                float v = acc[j] + bv;
                if (relu) v = fmaxf(v, 0.0f);
                Y[(size_t)grow * NC + col] = v;
            }
        }
    }
}

// ---------------------------------------------------------------------------
// Per-(node, head) attention logits: out[n*4+h] = sum_c lin[n,h*32+c]*att[h,c]
// ---------------------------------------------------------------------------
__global__ void att_kernel(const float* __restrict__ lin,
                           const float* __restrict__ attv,  // [4*32]
                           float* __restrict__ out, int total) {
    int g = blockIdx.x * blockDim.x + threadIdx.x;
    if (g >= total) return;
    int n = g >> 2, h = g & 3;
    const float* p = &lin[(size_t)n * HID + h * CHAN];
    const float* q = &attv[h * CHAN];
    float s = 0.f;
#pragma unroll
    for (int c = 0; c < CHAN; ++c) s += p[c] * q[c];
    out[g] = s;
}

// order-preserving float<->uint encoding for atomicMax on signed floats
__device__ __forceinline__ unsigned f2ord(float f) {
    unsigned u = __float_as_uint(f);
    return (u & 0x80000000u) ? ~u : (u | 0x80000000u);
}
__device__ __forceinline__ float ord2f(unsigned e) {
    unsigned u = (e & 0x80000000u) ? (e & 0x7fffffffu) : ~e;
    return __uint_as_float(u);
}
#define ORD_NEG_INF 0x007FFFFFu   // f2ord(-inf)

__device__ __forceinline__ float lrelu(float v) {
    return v > 0.f ? v : NEG_SLOPE * v;
}

__global__ void fill_u32_kernel(unsigned* __restrict__ p, unsigned v, size_t n) {
    size_t g = (size_t)blockIdx.x * blockDim.x + threadIdx.x;
    if (g < n) p[g] = v;
}

// edge pass 1: segment max (encoded)
__global__ void edge_max_kernel(const int* __restrict__ src, const int* __restrict__ dst,
                                const float* __restrict__ as, const float* __restrict__ ad,
                                unsigned* __restrict__ mEnc, int E_) {
    int e = blockIdx.x * blockDim.x + threadIdx.x;
    if (e >= E_) return;
    int s = src[e], d = dst[e];
#pragma unroll
    for (int h = 0; h < NHEAD; ++h) {
        float v = lrelu(as[s * NHEAD + h] + ad[d * NHEAD + h]);
        atomicMax(&mEnc[d * NHEAD + h], f2ord(v));
    }
}

// edge pass 2: ex = exp(e - m[dst]); denom += ex
__global__ void edge_expsum_kernel(const int* __restrict__ src, const int* __restrict__ dst,
                                   const float* __restrict__ as, const float* __restrict__ ad,
                                   const unsigned* __restrict__ mEnc,
                                   float* __restrict__ ex, float* __restrict__ den, int E_) {
    int e = blockIdx.x * blockDim.x + threadIdx.x;
    if (e >= E_) return;
    int s = src[e], d = dst[e];
#pragma unroll
    for (int h = 0; h < NHEAD; ++h) {
        float v = lrelu(as[s * NHEAD + h] + ad[d * NHEAD + h]);
        float m = ord2f(mEnc[d * NHEAD + h]);
        if (!__builtin_isfinite(m)) m = 0.f;     // matches reference semantics
        float x = expf(v - m);
        ex[(size_t)e * NHEAD + h] = x;
        atomicAdd(&den[d * NHEAD + h], x);
    }
}

// edge pass 3: out[dst, f] += alpha * hs[src, f]   (one thread per edge-feature)
__global__ void edge_scatter_kernel(const int* __restrict__ src, const int* __restrict__ dst,
                                    const float* __restrict__ ex, const float* __restrict__ den,
                                    const float* __restrict__ hs, float* __restrict__ out,
                                    size_t total) {
    size_t g = (size_t)blockIdx.x * blockDim.x + threadIdx.x;
    if (g >= total) return;
    int e = (int)(g >> 7);
    int f = (int)(g & 127);
    int h = f >> 5;
    int s = src[e], d = dst[e];
    float alpha = ex[(size_t)e * NHEAD + h] / fmaxf(den[d * NHEAD + h], 1e-16f);
    atomicAdd(&out[(size_t)d * HID + f], alpha * hs[(size_t)s * HID + f]);
}

// ---------------------------------------------------------------------------
// BatchNorm (eval-on-batch, population var): two-stage reduction + apply
// ---------------------------------------------------------------------------
__global__ void bn_stats_kernel(const float* __restrict__ x, float* __restrict__ stats,
                                int N) {
    __shared__ float s1[HID], s2[HID];
    int t = threadIdx.x;                 // 256 threads, 2 per channel
    if (t < HID) { s1[t] = 0.f; s2[t] = 0.f; }
    __syncthreads();
    int ch = t & 127, half = t >> 7;
    int rbeg = blockIdx.x * 256 + half;
    int rend = (blockIdx.x + 1) * 256;
    if (rend > N) rend = N;
    float a = 0.f, b = 0.f;
    for (int r = rbeg; r < rend; r += 2) {
        float v = x[(size_t)r * HID + ch];
        a += v; b += v * v;
    }
    atomicAdd(&s1[ch], a);
    atomicAdd(&s2[ch], b);
    __syncthreads();
    if (t < HID) {
        atomicAdd(&stats[t], s1[t]);
        atomicAdd(&stats[HID + t], s2[t]);
    }
}

__global__ void bn_apply_kernel(const float* __restrict__ x,
                                const float* __restrict__ stats,
                                const float* __restrict__ gamma,
                                const float* __restrict__ beta,
                                const float* __restrict__ res,   // nullable
                                float* __restrict__ y,
                                int N, int relu) {
    size_t g = (size_t)blockIdx.x * blockDim.x + threadIdx.x;
    size_t total = (size_t)N * HID;
    if (g >= total) return;
    int ch = (int)(g & 127);
    float inv_n = 1.0f / (float)N;
    float mu = stats[ch] * inv_n;
    float var = stats[HID + ch] * inv_n - mu * mu;
    float v = gamma[ch] * (x[g] - mu) * rsqrtf(var + BN_EPS) + beta[ch];
    if (relu) v = fmaxf(v, 0.0f);
    if (res) v += res[g];
    y[g] = v;
}

// ---------------------------------------------------------------------------

extern "C" void kernel_launch(void* const* d_in, const int* in_sizes, int n_in,
                              void* d_out, int out_size, void* d_ws, size_t ws_size,
                              hipStream_t stream) {
    (void)in_sizes; (void)n_in; (void)out_size; (void)ws_size;

    constexpr int NM = 100000, NA = 50000, E = 640000;

    const float* x_m  = (const float*)d_in[0];
    const float* x_a  = (const float*)d_in[1];
    const int*   e_src = (const int*)d_in[2];
    const int*   e_dst = (const int*)d_in[3];
    auto P = [&](int i) { return (const float*)d_in[4 + i]; };
    // param leaf order (dict insertion order):
    //   per layer l (base l*12): usesW,uses_as,uses_ad,uses_b, revW,rev_as,rev_ad,rev_b,
    //                            bn_m_g, bn_m_b, bn_a_g, bn_a_b
    //   36: proj_mW, 37: proj_mb, 38: proj_aW, 39: proj_ab
    //   40: fbn_m_g, 41: fbn_m_b, 42: fbn_a_g, 43: fbn_a_b
    //   44-47: head_m (W1,b1,W2,b2), 48-51: head_a

    // --- workspace carve ---
    float* wsf = (float*)d_ws;
    size_t off = 0;
    auto alloc = [&](size_t n) { float* p = wsf + off; off += n; return p; };
    float* bufAm = alloc((size_t)NM * HID);
    float* bufAa = alloc((size_t)NA * HID);
    float* bufBm = alloc((size_t)NM * HID);
    float* bufBa = alloc((size_t)NA * HID);
    float* hs    = alloc((size_t)NM * HID);   // lin(src) for current relation / fbn out
    float* tmp   = alloc((size_t)NM * HID);   // lin(dst) scratch / head hidden
    float* new_m = alloc((size_t)NM * HID);
    float* new_a = alloc((size_t)NA * HID);
    float* a_s   = alloc((size_t)NM * NHEAD);
    float* a_d   = alloc((size_t)NM * NHEAD);
    unsigned* mEnc = (unsigned*)alloc((size_t)NM * NHEAD);
    float* den   = alloc((size_t)NM * NHEAD);
    float* ex    = alloc((size_t)E * NHEAD);
    float* stats = alloc(2 * HID);
    _Float16* Wh = (_Float16*)alloc(256 * 128 / 2 + 64);  // transposed f16 weights

    auto cdiv = [](size_t a, size_t b) { return (unsigned)((a + b - 1) / b); };

    auto fill = [&](void* p, unsigned v, size_t n) {
        fill_u32_kernel<<<cdiv(n, 256), 256, 0, stream>>>((unsigned*)p, v, n);
    };
    auto lin128 = [&](const float* X, const float* W, const float* b, float* Y,
                      int N, int K, int relu) {
        w_to_f16t_kernel<<<cdiv((size_t)K * 128, 256), 256, 0, stream>>>(W, Wh, K, 128);
        lin_wmma_kernel<128><<<cdiv(N, 64), 256, 0, stream>>>(X, Wh, b, Y, N, K, relu);
    };
    auto lin64 = [&](const float* X, const float* W, const float* b, float* Y,
                     int N, int K, int relu) {
        w_to_f16t_kernel<<<cdiv((size_t)K * 64, 256), 256, 0, stream>>>(W, Wh, K, 64);
        lin_wmma_kernel<64><<<cdiv(N, 64), 128, 0, stream>>>(X, Wh, b, Y, N, K, relu);
    };

    // one GAT relation: out[Nd,128] (GAT bias folded out: BN is shift-invariant)
    auto run_gat = [&](const float* xs, int Ns, const float* xd, int Nd,
                       const int* es, const int* ed,
                       const float* W, const float* att_s, const float* att_d,
                       int K, float* out) {
        lin128(xs, W, nullptr, hs, Ns, K, 0);
        lin128(xd, W, nullptr, tmp, Nd, K, 0);
        att_kernel<<<cdiv((size_t)Ns * NHEAD, 256), 256, 0, stream>>>(hs, att_s, a_s, Ns * NHEAD);
        att_kernel<<<cdiv((size_t)Nd * NHEAD, 256), 256, 0, stream>>>(tmp, att_d, a_d, Nd * NHEAD);
        fill(mEnc, ORD_NEG_INF, (size_t)Nd * NHEAD);
        fill(den, 0u, (size_t)Nd * NHEAD);
        fill(out, 0u, (size_t)Nd * HID);
        edge_max_kernel<<<cdiv(E, 256), 256, 0, stream>>>(es, ed, a_s, a_d, mEnc, E);
        edge_expsum_kernel<<<cdiv(E, 256), 256, 0, stream>>>(es, ed, a_s, a_d, mEnc, ex, den, E);
        edge_scatter_kernel<<<cdiv((size_t)E * HID, 256), 256, 0, stream>>>(
            es, ed, ex, den, hs, out, (size_t)E * HID);
    };

    const float* cm = x_m;
    const float* ca = x_a;
    int K = 256;

    for (int l = 0; l < 3; ++l) {
        int pb = l * 12;
        // ('mashup','uses','api'): messages mashup -> api
        run_gat(cm, NM, ca, NA, e_src, e_dst, P(pb + 0), P(pb + 1), P(pb + 2), K, new_a);
        // reverse: api -> mashup
        run_gat(ca, NA, cm, NM, e_dst, e_src, P(pb + 4), P(pb + 5), P(pb + 6), K, new_m);

        float* nm = (l % 2 == 0) ? bufAm : bufBm;
        float* na = (l % 2 == 0) ? bufAa : bufBa;
        const float *resm, *resa;
        if (l == 0) {
            lin128(cm, P(36), P(37), nm, NM, 256, 0);   // proj_m residual
            lin128(ca, P(38), P(39), na, NA, 256, 0);   // proj_a residual
            resm = nm; resa = na;
        } else {
            resm = cm; resa = ca;
        }

        // new_m -> relu(bn(new_m)) + res -> nm
        fill(stats, 0u, 2 * HID);
        bn_stats_kernel<<<cdiv(NM, 256), 256, 0, stream>>>(new_m, stats, NM);
        bn_apply_kernel<<<cdiv((size_t)NM * HID, 256), 256, 0, stream>>>(
            new_m, stats, P(pb + 8), P(pb + 9), resm, nm, NM, 1);
        // new_a -> relu(bn(new_a)) + res -> na
        fill(stats, 0u, 2 * HID);
        bn_stats_kernel<<<cdiv(NA, 256), 256, 0, stream>>>(new_a, stats, NA);
        bn_apply_kernel<<<cdiv((size_t)NA * HID, 256), 256, 0, stream>>>(
            new_a, stats, P(pb + 10), P(pb + 11), resa, na, NA, 1);

        cm = nm; ca = na; K = 128;
    }

    float* out_m = (float*)d_out;
    float* out_a = (float*)d_out + (size_t)NM * 64;

    // final BN + MLP head, mashup
    fill(stats, 0u, 2 * HID);
    bn_stats_kernel<<<cdiv(NM, 256), 256, 0, stream>>>(cm, stats, NM);
    bn_apply_kernel<<<cdiv((size_t)NM * HID, 256), 256, 0, stream>>>(
        cm, stats, P(40), P(41), nullptr, hs, NM, 0);
    lin64(hs, P(44), P(45), tmp, NM, 128, 1);
    lin64(tmp, P(46), P(47), out_m, NM, 64, 0);

    // final BN + MLP head, api
    fill(stats, 0u, 2 * HID);
    bn_stats_kernel<<<cdiv(NA, 256), 256, 0, stream>>>(ca, stats, NA);
    bn_apply_kernel<<<cdiv((size_t)NA * HID, 256), 256, 0, stream>>>(
        ca, stats, P(42), P(43), nullptr, hs, NA, 0);
    lin64(hs, P(48), P(49), tmp, NA, 128, 1);
    lin64(tmp, P(50), P(51), out_a, NA, 64, 0);
}
